// Self_Attn_suvey_mask_58609123721847
// MI455X (gfx1250) — matrix operand order repaired and checked by reading, not compiled, not run on hardware
//
#include <hip/hip_runtime.h>
#include <math.h>

typedef __attribute__((ext_vector_type(16))) _Float16 v16h;
typedef __attribute__((ext_vector_type(8)))  _Float16 v8h;
typedef __attribute__((ext_vector_type(4)))  _Float16 v4h;
typedef __attribute__((ext_vector_type(2)))  _Float16 v2h;
typedef __attribute__((ext_vector_type(8)))  float    v8f;

#define LSEQ 2048
#define CCH  512
#define DQK  64
#define OTOT 640   // 64 q + 64 k + 512 v rows in fused projection

__device__ inline v16h frag_cat(v8h lo, v8h hi) {
    return __builtin_shufflevector(lo, hi, 0,1,2,3,4,5,6,7,8,9,10,11,12,13,14,15);
}

// A fragment (16 rows x 32 K slice), LDS row-major [row][K], stride ldk halves.
__device__ inline v16h load_fragA(const _Float16* lds, int lane, int mbase, int kbase, int ldk) {
    int row  = mbase + (lane & 15);
    int koff = (lane & 16) ? 8 : 0;
    const _Float16* p = lds + row * ldk + kbase + koff;
    v8h lo = *(const v8h*)(p);
    v8h hi = *(const v8h*)(p + 16);
    return frag_cat(lo, hi);
}

// B fragment (32 K x 16 cols), LDS stored [col][K], stride ldk halves.
__device__ inline v16h load_fragB(const _Float16* lds, int lane, int nbase, int kbase, int ldk) {
    int col  = nbase + (lane & 15);
    int koff = (lane & 16) ? 8 : 0;   // 0 or 8
    const _Float16* p = lds + col * ldk + kbase + koff;
    v8h lo = *(const v8h*)(p);
    v8h hi = *(const v8h*)(p + 16);
    return frag_cat(lo, hi);
}

__device__ inline v8f wmma16(v16h a, v16h b, v8f c) {
    return __builtin_amdgcn_wmma_f32_16x16x32_f16(false, a, false, b, (short)0, c, false, false);
}

__device__ inline v2h cvt2(float lo, float hi) {
    v2h r; r.x = (_Float16)lo; r.y = (_Float16)hi; return r;
}

// ---------------------------------------------------------------------------
// Kernel A: fused QKV projection.  P[b][o][l] = Wcat[o][:] . x[b][:][l] + bcat[o]
// Block tile 64(M=o) x 128(N=l); block-uniform W select; register-prefetch
// pipeline; packed b32 LDS staging stores.
// ---------------------------------------------------------------------------
__global__ __launch_bounds__(256) void qkv_proj_kernel(
    const float* __restrict__ x,
    const float* __restrict__ Wq, const float* __restrict__ bq,
    const float* __restrict__ Wk, const float* __restrict__ bk,
    const float* __restrict__ Wv, const float* __restrict__ bv,
    _Float16* __restrict__ Ph)
{
    __shared__ __align__(16) _Float16 smA[64 * 32];    // W tile  [o][k]
    __shared__ __align__(16) _Float16 smB[128 * 32];   // x tile  [l][k]
    const int b       = blockIdx.z;
    const int rowBase = blockIdx.y * 64;    // o
    const int colBase = blockIdx.x * 128;   // l
    const int t = threadIdx.x, lane = t & 31, w = t >> 5;
    const int Mi = w & 3, Nq = (w >> 2) * 4;

    const float* Wsel; const float* bsel;
    if (rowBase == 0)       { Wsel = Wq;                         bsel = bq; }
    else if (rowBase == 64) { Wsel = Wk;                         bsel = bk; }
    else                    { Wsel = Wv + (rowBase - 128) * CCH; bsel = bv + (rowBase - 128); }

    // A fill: rows (t>>4)+16i, k-pair 2*(t&15).  B fill: col t&127, k-pairs 2*(t>>7)+4i.
    const int ar = t >> 4, akp = t & 15;
    const float* aBase = Wsel + ar * CCH + 2 * akp;
    const int bl = t & 127, bkp = t >> 7;
    const float* bBase = x + (size_t)b * CCH * LSEQ + (size_t)(2 * bkp) * LSEQ + colBase + bl;

    float2 aPre[4]; float bLo[8], bHi[8];
    auto prefetch = [&](int k0) {
#pragma unroll
        for (int i = 0; i < 4; ++i) aPre[i] = *(const float2*)(aBase + i * 16 * CCH + k0);
#pragma unroll
        for (int i = 0; i < 8; ++i) {
            bLo[i] = bBase[(size_t)(k0 + 4 * i) * LSEQ];
            bHi[i] = bBase[(size_t)(k0 + 4 * i) * LSEQ + LSEQ];
        }
    };
    prefetch(0);

    v8f acc[4] = {};
    for (int k0 = 0; k0 < CCH; k0 += 32) {
        __syncthreads();
#pragma unroll
        for (int i = 0; i < 4; ++i)
            *(v2h*)&smA[(ar + 16 * i) * 32 + 2 * akp] = cvt2(aPre[i].x, aPre[i].y);
#pragma unroll
        for (int i = 0; i < 8; ++i)
            *(v2h*)&smB[bl * 32 + 2 * bkp + 4 * i] = cvt2(bLo[i], bHi[i]);
        __syncthreads();
        if (k0 + 32 < CCH) prefetch(k0 + 32);      // overlap next-tile loads with WMMA

        v16h a = load_fragA(smA, lane, Mi * 16, 0, 32);
        v16h bf[4];
#pragma unroll
        for (int j = 0; j < 4; ++j) bf[j] = load_fragB(smB, lane, (Nq + j) * 16, 0, 32);
#pragma unroll
        for (int j = 0; j < 4; ++j) acc[j] = wmma16(a, bf[j], acc[j]);
    }
    const int oLoc0 = Mi * 16 + ((lane & 16) ? 8 : 0);
    const int col   = colBase + Nq * 16 + (lane & 15);
#pragma unroll
    for (int r = 0; r < 8; ++r) {
        float bias  = bsel[oLoc0 + r];
        size_t base = ((size_t)b * OTOT + rowBase + oLoc0 + r) * LSEQ + col;
#pragma unroll
        for (int j = 0; j < 4; ++j)
            Ph[base + j * 16] = (_Float16)(acc[j][r] + bias);
    }
}

// ---------------------------------------------------------------------------
// Kernel B: energy[b][l][m] = q[b][l][:] . k[b][:][m]   (64 l x 128 m tiles,
// lower triangle only; masked entries -> -inf; fully-masked tiles skipped).
// ---------------------------------------------------------------------------
__global__ __launch_bounds__(256) void energy_kernel(
    const _Float16* __restrict__ Ph,
    const unsigned char* __restrict__ dmask,
    float* __restrict__ att)
{
    const int b     = blockIdx.z;
    const int lBase = blockIdx.y * 64;
    const int mBase = blockIdx.x * 128;
    if (mBase > lBase + 63) return;   // entirely causal-masked: never read by softmax
    __shared__ __align__(16) _Float16 smQ[64 * 64];    // [l][d]
    __shared__ __align__(16) _Float16 smK[128 * 64];   // [m][d]
    const int t = threadIdx.x, lane = t & 31, w = t >> 5;
    const _Float16* Pq = Ph + (size_t)b * OTOT * LSEQ;   // rows 0..63  (q)
    const _Float16* Pk = Pq + (size_t)DQK * LSEQ;        // rows 64..127 (k)

    // Stage with 64-bit (4-half) loads; transpose into [seq][d] in LDS.
#pragma unroll
    for (int i = 0; i < 4; ++i) {                  // 1024 quads for smQ
        int idx = i * 256 + t;
        int d = idx >> 4, jq = idx & 15;           // seq = 4*jq
        v4h q4 = *(const v4h*)(Pq + (size_t)d * LSEQ + lBase + 4 * jq);
#pragma unroll
        for (int s = 0; s < 4; ++s) smQ[(4 * jq + s) * 64 + d] = q4[s];
    }
#pragma unroll
    for (int i = 0; i < 8; ++i) {                  // 2048 quads for smK
        int idx = i * 256 + t;
        int d = idx >> 5, mq = idx & 31;           // m = 4*mq
        v4h k4 = *(const v4h*)(Pk + (size_t)d * LSEQ + mBase + 4 * mq);
#pragma unroll
        for (int s = 0; s < 4; ++s) smK[(4 * mq + s) * 64 + d] = k4[s];
    }
    __syncthreads();
    const int Mi = w & 3, Nq = (w >> 2) * 4;
    v8f acc[4] = {};
#pragma unroll
    for (int k0 = 0; k0 < DQK; k0 += 32) {
        v16h a = load_fragA(smQ, lane, Mi * 16, k0, 64);
        v16h bf[4];
#pragma unroll
        for (int j = 0; j < 4; ++j) bf[j] = load_fragB(smK, lane, (Nq + j) * 16, k0, 64);
#pragma unroll
        for (int j = 0; j < 4; ++j) acc[j] = wmma16(a, bf[j], acc[j]);
    }
    const int l0 = lBase + Mi * 16 + ((lane & 16) ? 8 : 0);
    const int m0 = mBase + Nq * 16 + (lane & 15);
#pragma unroll
    for (int r = 0; r < 8; ++r) {
        int l = l0 + r;
        size_t rowoff = ((size_t)b * LSEQ + l) * LSEQ;
#pragma unroll
        for (int j = 0; j < 4; ++j) {
            int m = m0 + j * 16;
            bool msk = (m > l) || (dmask[rowoff + m] != 0);
            att[rowoff + m] = msk ? -INFINITY : acc[j][r];
        }
    }
}

// ---------------------------------------------------------------------------
// Kernel C: in-place row softmax (mask applied by index; vectorized float4).
// ---------------------------------------------------------------------------
__global__ __launch_bounds__(256) void softmax_kernel(
    const unsigned char* __restrict__ dmask,
    float* __restrict__ att)
{
    const int l = blockIdx.x, b = blockIdx.y;
    const size_t rowoff = ((size_t)b * LSEQ + l) * LSEQ;
    const int t = threadIdx.x;
    __shared__ float red[256];

    float e[8]; bool valid[8];
    float mx = -INFINITY;
#pragma unroll
    for (int i = 0; i < 2; ++i) {
        int m = (i * 256 + t) * 4;
        float4 ev = *(const float4*)(att + rowoff + m);
        uchar4 dm = *(const uchar4*)(dmask + rowoff + m);
        float vv[4] = {ev.x, ev.y, ev.z, ev.w};
        unsigned char dd[4] = {dm.x, dm.y, dm.z, dm.w};
#pragma unroll
        for (int s = 0; s < 4; ++s) {
            bool v = (m + s <= l) && (dd[s] == 0);
            float val = v ? vv[s] : -INFINITY;
            e[i * 4 + s] = val; valid[i * 4 + s] = v;
            mx = fmaxf(mx, val);
        }
    }
    red[t] = mx; __syncthreads();
    for (int s = 128; s > 0; s >>= 1) { if (t < s) red[t] = fmaxf(red[t], red[t + s]); __syncthreads(); }
    mx = red[0]; __syncthreads();

    float p[8]; float sum = 0.f;
#pragma unroll
    for (int i = 0; i < 8; ++i) {
        float pv = valid[i] ? __expf(e[i] - mx) : 0.f;
        p[i] = pv; sum += pv;
    }
    red[t] = sum; __syncthreads();
    for (int s = 128; s > 0; s >>= 1) { if (t < s) red[t] += red[t + s]; __syncthreads(); }
    float inv = 1.0f / red[0];
#pragma unroll
    for (int i = 0; i < 2; ++i) {
        int m = (i * 256 + t) * 4;
        float4 ov = make_float4(p[i*4+0]*inv, p[i*4+1]*inv, p[i*4+2]*inv, p[i*4+3]*inv);
        *(float4*)(att + rowoff + m) = ov;
    }
}

// ---------------------------------------------------------------------------
// Kernel D: out[b][c][m] = sum_l v[b][c][l] * att[b][m][l]
// Block tile 64(M=c) x 128(N=m); K loop stops at the diagonal; register
// prefetch pipeline; packed b64/b32 LDS staging.
// ---------------------------------------------------------------------------
__global__ __launch_bounds__(256) void out_kernel(
    const _Float16* __restrict__ Ph,
    const float* __restrict__ att,
    float* __restrict__ out)
{
    __shared__ __align__(16) _Float16 smV[64 * 32];    // [c][k]
    __shared__ __align__(16) _Float16 smT[128 * 32];   // [m][k]
    const int b     = blockIdx.z;
    const int cBase = blockIdx.y * 64;
    const int mBase = blockIdx.x * 128;
    const int t = threadIdx.x, lane = t & 31, w = t >> 5;
    const int Mi = w & 3, Nq = (w >> 2) * 4;

    // V fill: rows (t>>3)+32i, k-quad 4*(t&7).  T fill: rows (t>>4)+16i, k-pair 2*(t&15).
    const int vrr = t >> 3, vkq = t & 7;
    const _Float16* vBase = Ph + ((size_t)b * OTOT + 128 + cBase + vrr) * LSEQ + 4 * vkq;
    const int trr = t >> 4, tkp = t & 15;
    const float* tBase = att + ((size_t)b * LSEQ + mBase + trr) * LSEQ + 2 * tkp;

    v4h vPre[2]; float2 tPre[8];
    auto prefetch = [&](int k0) {
#pragma unroll
        for (int i = 0; i < 2; ++i) vPre[i] = *(const v4h*)(vBase + (size_t)i * 32 * LSEQ + k0);
#pragma unroll
        for (int i = 0; i < 8; ++i) tPre[i] = *(const float2*)(tBase + (size_t)i * 16 * LSEQ + k0);
    };
    prefetch(0);

    const int kEnd = mBase + 128;   // att is exactly 0 beyond the diagonal
    v8f acc[4] = {};
    for (int k0 = 0; k0 < kEnd; k0 += 32) {
        __syncthreads();
#pragma unroll
        for (int i = 0; i < 2; ++i)
            *(v4h*)&smV[(vrr + 32 * i) * 32 + 4 * vkq] = vPre[i];
#pragma unroll
        for (int i = 0; i < 8; ++i)
            *(v2h*)&smT[(trr + 16 * i) * 32 + 2 * tkp] = cvt2(tPre[i].x, tPre[i].y);
        __syncthreads();
        if (k0 + 32 < kEnd) prefetch(k0 + 32);     // overlap next-tile loads with WMMA

        v16h a = load_fragA(smV, lane, Mi * 16, 0, 32);
        v16h bf[4];
#pragma unroll
        for (int j = 0; j < 4; ++j) bf[j] = load_fragB(smT, lane, (Nq + j) * 16, 0, 32);
#pragma unroll
        for (int j = 0; j < 4; ++j) acc[j] = wmma16(a, bf[j], acc[j]);
    }
    const int c0 = cBase + Mi * 16 + ((lane & 16) ? 8 : 0);
    const int m0 = mBase + Nq * 16 + (lane & 15);
#pragma unroll
    for (int r = 0; r < 8; ++r) {
        size_t base = ((size_t)b * CCH + c0 + r) * LSEQ + m0;
#pragma unroll
        for (int j = 0; j < 4; ++j)
            out[base + j * 16] = acc[j][r];
    }
}

extern "C" void kernel_launch(void* const* d_in, const int* in_sizes, int n_in,
                              void* d_out, int out_size, void* d_ws, size_t ws_size,
                              hipStream_t stream) {
    const float*         x     = (const float*)d_in[0];
    const unsigned char* dmask = (const unsigned char*)d_in[1];   // bool (B,L,L)
    const float* Wq = (const float*)d_in[2];
    const float* bq = (const float*)d_in[3];
    const float* Wk = (const float*)d_in[4];
    const float* bk = (const float*)d_in[5];
    const float* Wv = (const float*)d_in[6];
    const float* bv = (const float*)d_in[7];

    float* out = (float*)d_out;
    float* att = out + (size_t)8 * CCH * LSEQ;      // second tuple output
    _Float16* Ph = (_Float16*)d_ws;                 // (B, 640, 2048) f16 = 21 MB

    dim3 blk(256);
    qkv_proj_kernel<<<dim3(LSEQ / 128, OTOT / 64, 8), blk, 0, stream>>>(
        x, Wq, bq, Wk, bk, Wv, bv, Ph);
    energy_kernel<<<dim3(LSEQ / 128, LSEQ / 64, 8), blk, 0, stream>>>(Ph, dmask, att);
    softmax_kernel<<<dim3(LSEQ, 8), blk, 0, stream>>>(dmask, att);
    out_kernel<<<dim3(LSEQ / 128, CCH / 64, 8), blk, 0, stream>>>(Ph, att, out);
}
